// NaiveLSTMCell_16183436772221
// MI455X (gfx1250) — compile-verified, matrix operand order checked by reading, not run on hardware
//
#include <hip/hip_runtime.h>
#include <hip/hip_bf16.h>

// ---------------------------------------------------------------------------
// Fused LSTM cell for MI455X (gfx1250, wave32, WMMA).
//   gates = x @ Wx[g] + h @ Wh[g] + b[g]   (g = i, f, c, o)
//   cy = sigmoid(f)*c_x + sigmoid(i)*tanh(c);  hy = sigmoid(o)*tanh(cy)
// Compute-bound (~950 flop/byte) -> bf16 WMMA with f32 accumulate.
// Staging: GLOBAL_LOAD_ASYNC_TO_LDS_B128 (ASYNCcnt), double-buffered LDS,
// compile-time buffer index (k-loop unrolled 2x).  sched_barrier(0) splits
// fragment loads from WMMAs so all 20 ds_load_b128 issue before the 16 WMMAs.
// Epilogue uses gfx1250 v_tanh_f32 (TRANS) for tanh / tanh-based sigmoid.
// ---------------------------------------------------------------------------

typedef __attribute__((ext_vector_type(16))) __bf16 v16bf;
typedef __attribute__((ext_vector_type(8)))  __bf16 bf16x8;
typedef __attribute__((ext_vector_type(8)))  float  v8f;

#define LSTM_BATCH 4096
#define LSTM_N     2048   // hidden size
#define LSTM_K     2048   // input size (== hidden here)
#define BM 128            // block tile M (batch)
#define BN 64             // block tile N (hidden)
#define BK 32             // k-step = one WMMA K
#define LDT 40            // padded LDS row stride in ushort (80 B, 16B-aligned)

__device__ __forceinline__ unsigned short f2bf(float f) {
  unsigned int u = __float_as_uint(f);
  u += 0x7FFFu + ((u >> 16) & 1u);           // round-to-nearest-even
  return (unsigned short)(u >> 16);
}

// ---- pre-pass: f32 -> bf16 elementwise (x, h) ------------------------------
__global__ void cvt_bf16(const float* __restrict__ src,
                         unsigned short* __restrict__ dst) {
  const size_t i = ((size_t)blockIdx.x * blockDim.x + threadIdx.x) * 4;
  float4 v = *(const float4*)(src + i);
  union { unsigned short u[4]; uint2 q; } o;
  o.u[0] = f2bf(v.x); o.u[1] = f2bf(v.y);
  o.u[2] = f2bf(v.z); o.u[3] = f2bf(v.w);
  *(uint2*)(dst + i) = o.q;
}

// ---- pre-pass: W[k][n] f32 -> Wt[n][k] bf16 (tiled transpose) --------------
__global__ void xpose_bf16(const float* __restrict__ W,
                           unsigned short* __restrict__ Wt) {
  __shared__ float t[32][33];
  const int n0 = blockIdx.x * 32, k0 = blockIdx.y * 32;
  const int tx = threadIdx.x, ty = threadIdx.y;   // 32 x 8
#pragma unroll
  for (int i = 0; i < 4; ++i)
    t[ty + 8 * i][tx] = W[(size_t)(k0 + ty + 8 * i) * LSTM_N + n0 + tx];
  __syncthreads();
#pragma unroll
  for (int i = 0; i < 4; ++i)
    Wt[(size_t)(n0 + ty + 8 * i) * LSTM_K + k0 + tx] = f2bf(t[tx][ty + 8 * i]);
}

// ---- CDNA5 async global->LDS copy (ASYNCcnt), 16 B per lane ----------------
__device__ __forceinline__ unsigned lds_off(const void* p) {
  // generic LDS address carries the LDS byte offset in bits [31:0]
  return (unsigned)(unsigned long long)p;
}
__device__ __forceinline__ void async_b128(unsigned dst_lds, const void* src) {
  asm volatile("global_load_async_to_lds_b128 %0, %1, off"
               :: "v"(dst_lds), "v"(src) : "memory");
}
__device__ __forceinline__ void wait_async0() {
  asm volatile("s_wait_asynccnt 0" ::: "memory");
}

// ---- gfx1250 hardware tanh (TRANS op; 1-op gap before use per ISA 7.4) -----
__device__ __forceinline__ float fast_tanh(float x) {
  float r;
  asm volatile("v_tanh_f32 %0, %1\n\tv_nop" : "=v"(r) : "v"(x));
  return r;
}
__device__ __forceinline__ float fast_sigmoid(float x) {
  return fmaf(0.5f, fast_tanh(0.5f * x), 0.5f);
}

// ---- load one 16x32 bf16 WMMA fragment from an LDS tile --------------------
// ISA layout (CDNA5 ISA): lanes 0-15 hold M(or N)=lane, K=0..7 & 16..23;
// lanes 16-31 hold K=8..15 & 24..31.  Tile is [16 rows][LDT] bf16, row-major.
__device__ __forceinline__ v16bf ld_frag(const unsigned short* base, int lane) {
  const int m  = lane & 15;
  const int kb = (lane >> 4) << 3;                // 0 or 8
  const __bf16* p = (const __bf16*)(base + m * LDT + kb);
  bf16x8 lo = *(const bf16x8*)(p);                // K = kb .. kb+7
  bf16x8 hi = *(const bf16x8*)(p + 16);           // K = kb+16 .. kb+23
  return __builtin_shufflevector(lo, hi, 0, 1, 2, 3, 4, 5, 6, 7,
                                         8, 9, 10, 11, 12, 13, 14, 15);
}

// ---- fused LSTM main kernel ------------------------------------------------
__global__ __launch_bounds__(256)
void lstm_wmma(const unsigned short* __restrict__ xb,
               const unsigned short* __restrict__ hb,
               const unsigned short* __restrict__ Wx,   // 4 gates, each [N][K]
               const unsigned short* __restrict__ Wh,   // 4 gates, each [N][K]
               const float* __restrict__ b_i, const float* __restrict__ b_f,
               const float* __restrict__ b_c, const float* __restrict__ b_o,
               const float* __restrict__ c_x,
               float* __restrict__ hy, float* __restrict__ cy) {
  // double-buffered tiles: 2 x (A 128x32 + 4 x B 64x32) bf16, padded rows
  __shared__ __align__(16) unsigned short As[2][BM * LDT];
  __shared__ __align__(16) unsigned short Bs[2][4][BN * LDT];

  const int tid  = threadIdx.x;
  const int lane = tid & 31;
  const int wave = tid >> 5;           // 8 waves
  const int wm = wave >> 1;            // 0..3  (32-row slice of M)
  const int wn = wave & 1;             // 0..1  (32-col slice of N)
  const int bM = blockIdx.x * BM;
  const int bN = blockIdx.y * BN;

  const int sr = tid >> 2;             // staging row 0..63
  const int sc = (tid & 3) * 8;        // staging col (bf16 elems)

  // issue 6 async b128 copies for k-block kb into LDS buffer `buf`
  auto stage = [&](int kb, int buf) {
    const unsigned short* Ab = (kb < 64) ? xb : hb;
    const unsigned short* Wb = (kb < 64) ? Wx : Wh;
    const int k0 = (kb & 63) * BK;
    async_b128(lds_off(&As[buf][sr * LDT + sc]),
               &Ab[(size_t)(bM + sr) * LSTM_K + k0 + sc]);
    async_b128(lds_off(&As[buf][(sr + 64) * LDT + sc]),
               &Ab[(size_t)(bM + sr + 64) * LSTM_K + k0 + sc]);
#pragma unroll
    for (int g = 0; g < 4; ++g)
      async_b128(lds_off(&Bs[buf][g][sr * LDT + sc]),
                 &Wb[(size_t)g * LSTM_N * LSTM_K +
                     (size_t)(bN + sr) * LSTM_K + k0 + sc]);
  };

  v8f acc[4][2][2] = {};               // [gate][tm][tn], f32 accum

  // one k-step on compile-time buffer `cur`: 20 ds_loads, then 16 WMMAs
  auto kstep = [&](int cur) {
    const v16bf a0 = ld_frag(&As[cur][(wm * 32 + 0)  * LDT], lane);
    const v16bf a1 = ld_frag(&As[cur][(wm * 32 + 16) * LDT], lane);
    v16bf bfr[4][2];
#pragma unroll
    for (int g = 0; g < 4; ++g) {
      bfr[g][0] = ld_frag(&Bs[cur][g][(wn * 32 + 0)  * LDT], lane);
      bfr[g][1] = ld_frag(&Bs[cur][g][(wn * 32 + 16) * LDT], lane);
    }
    __builtin_amdgcn_sched_barrier(0);   // all loads issue before any WMMA
#pragma unroll
    for (int g = 0; g < 4; ++g) {
      acc[g][0][0] = __builtin_amdgcn_wmma_f32_16x16x32_bf16(
          false, a0, false, bfr[g][0], (short)0, acc[g][0][0], false, false);
      acc[g][0][1] = __builtin_amdgcn_wmma_f32_16x16x32_bf16(
          false, a0, false, bfr[g][1], (short)0, acc[g][0][1], false, false);
      acc[g][1][0] = __builtin_amdgcn_wmma_f32_16x16x32_bf16(
          false, a1, false, bfr[g][0], (short)0, acc[g][1][0], false, false);
      acc[g][1][1] = __builtin_amdgcn_wmma_f32_16x16x32_bf16(
          false, a1, false, bfr[g][1], (short)0, acc[g][1][1], false, false);
    }
    wait_async0();      // next buffer fully staged (this wave's copies)
    __syncthreads();    // ... and everyone else's; LDS reads above drained
  };

  stage(0, 0);
  wait_async0();
  __syncthreads();

#pragma unroll 1
  for (int it = 0; it < 64; ++it) {
    const int kb = it << 1;
    stage(kb + 1, 1);
    kstep(0);
    if (kb + 2 < 128) stage(kb + 2, 0);
    kstep(1);
  }

  // ---- epilogue: bias + activations + cell update, straight to HBM --------
  // C/D layout: element (vgpr vr, lane) = row M = vr + (lane>=16 ? 8:0),
  //             col N = lane & 15.
  const int lr = (lane >> 4) << 3;
  const int lc = lane & 15;
#pragma unroll
  for (int tn = 0; tn < 2; ++tn) {
    const int col = bN + wn * 32 + tn * 16 + lc;
    const float vbi = b_i[col], vbff = b_f[col];
    const float vbc = b_c[col], vbo = b_o[col];
#pragma unroll
    for (int tm = 0; tm < 2; ++tm) {
#pragma unroll
      for (int vr = 0; vr < 8; ++vr) {
        const int row = bM + wm * 32 + tm * 16 + lr + vr;
        const size_t idx = (size_t)row * LSTM_N + col;
        const float ig = fast_sigmoid(acc[0][tm][tn][vr] + vbi);
        const float fg = fast_sigmoid(acc[1][tm][tn][vr] + vbff);
        const float ct = fast_tanh(acc[2][tm][tn][vr] + vbc);
        const float og = fast_sigmoid(acc[3][tm][tn][vr] + vbo);
        const float cv = fg * c_x[idx] + ig * ct;
        hy[idx] = og * fast_tanh(cv);
        cy[idx] = cv;
      }
    }
  }
}

// ---------------------------------------------------------------------------
extern "C" void kernel_launch(void* const* d_in, const int* in_sizes, int n_in,
                              void* d_out, int out_size, void* d_ws,
                              size_t ws_size, hipStream_t stream) {
  const float* x    = (const float*)d_in[0];
  const float* h_x  = (const float*)d_in[1];
  const float* c_x  = (const float*)d_in[2];
  const float* W_xi = (const float*)d_in[3];
  const float* W_hi = (const float*)d_in[4];
  const float* b_i  = (const float*)d_in[5];
  const float* W_xf = (const float*)d_in[6];
  const float* W_hf = (const float*)d_in[7];
  const float* b_f  = (const float*)d_in[8];
  const float* W_xc = (const float*)d_in[9];
  const float* W_hc = (const float*)d_in[10];
  const float* b_c  = (const float*)d_in[11];
  const float* W_xo = (const float*)d_in[12];
  const float* W_ho = (const float*)d_in[13];
  const float* b_o  = (const float*)d_in[14];

  const size_t NB = (size_t)LSTM_BATCH * LSTM_K;   // 8Mi elems
  const size_t NW = (size_t)LSTM_N * LSTM_K;       // 4Mi elems
  // workspace: xb(16MB) hb(16MB) Wx(32MB) Wh(32MB) = 96MB bf16
  unsigned short* xb = (unsigned short*)d_ws;
  unsigned short* hb = xb + NB;
  unsigned short* Wx = hb + NB;
  unsigned short* Wh = Wx + 4 * NW;

  cvt_bf16<<<(unsigned)(NB / 4 / 256), 256, 0, stream>>>(x, xb);
  cvt_bf16<<<(unsigned)(NB / 4 / 256), 256, 0, stream>>>(h_x, hb);

  dim3 tb(32, 8), tg(LSTM_N / 32, LSTM_K / 32);
  xpose_bf16<<<tg, tb, 0, stream>>>(W_xi, Wx + 0 * NW);
  xpose_bf16<<<tg, tb, 0, stream>>>(W_xf, Wx + 1 * NW);
  xpose_bf16<<<tg, tb, 0, stream>>>(W_xc, Wx + 2 * NW);
  xpose_bf16<<<tg, tb, 0, stream>>>(W_xo, Wx + 3 * NW);
  xpose_bf16<<<tg, tb, 0, stream>>>(W_hi, Wh + 0 * NW);
  xpose_bf16<<<tg, tb, 0, stream>>>(W_hf, Wh + 1 * NW);
  xpose_bf16<<<tg, tb, 0, stream>>>(W_hc, Wh + 2 * NW);
  xpose_bf16<<<tg, tb, 0, stream>>>(W_ho, Wh + 3 * NW);

  float* hy = (float*)d_out;
  float* cy = hy + (size_t)LSTM_BATCH * LSTM_N;
  lstm_wmma<<<dim3(LSTM_BATCH / BM, LSTM_N / BN), 256, 0, stream>>>(
      xb, hb, Wx, Wh, b_i, b_f, b_c, b_o, c_x, hy, cy);
}
